// PointNet2SemSeg_19061064860310
// MI455X (gfx1250) — matrix operand order repaired and checked by reading, not compiled, not run on hardware
//
#include <hip/hip_runtime.h>

#define BATCH 4
#define NPTS 16384

typedef __attribute__((ext_vector_type(16))) _Float16 v16h;
typedef __attribute__((ext_vector_type(8)))  _Float16 v8h;
typedef __attribute__((ext_vector_type(8)))  float    v8f;

// ---------------------------------------------------------------------------
// Farthest point sampling: one workgroup per batch. dist[] lives in LDS
// (16384 floats = 64KB, well inside the 320KB/WGP budget). Each of NPOINT
// sequential steps does a block-wide argmax (wave32 shuffle + LDS reduce).
// ---------------------------------------------------------------------------
template<int N, int NPOINT, int THREADS>
__global__ __launch_bounds__(THREADS)
void fps_kernel(const float* __restrict__ xyz, float* __restrict__ new_xyz) {
  __shared__ float dist[N];
  __shared__ float red_d[THREADS / 32];
  __shared__ int   red_i[THREADS / 32];
  const int b = blockIdx.x;
  const float* px = xyz + (size_t)b * N * 3;
  float* out = new_xyz + (size_t)b * NPOINT * 3;
  const int t = threadIdx.x;
  for (int i = t; i < N; i += THREADS) dist[i] = 1e10f;
  __syncthreads();
  int far = 0;
  for (int it = 0; it < NPOINT; ++it) {
    float fx = px[far * 3 + 0], fy = px[far * 3 + 1], fz = px[far * 3 + 2];
    if (t == 0) { out[it * 3 + 0] = fx; out[it * 3 + 1] = fy; out[it * 3 + 2] = fz; }
    float bd = -1.0f; int bi = N - 1;
    for (int i = t; i < N; i += THREADS) {
      float dx = px[i * 3 + 0] - fx, dy = px[i * 3 + 1] - fy, dz = px[i * 3 + 2] - fz;
      float d = dx * dx + dy * dy + dz * dz;
      float nd = fminf(dist[i], d);
      dist[i] = nd;
      if (nd > bd || (nd == bd && i < bi)) { bd = nd; bi = i; }
    }
    // wave32 reduction (argmax, ties -> lowest index, matching jnp.argmax)
    for (int off = 16; off > 0; off >>= 1) {
      float od = __shfl_down(bd, off, 32);
      int   oi = __shfl_down(bi, off, 32);
      if (od > bd || (od == bd && oi < bi)) { bd = od; bi = oi; }
    }
    if ((t & 31) == 0) { red_d[t >> 5] = bd; red_i[t >> 5] = bi; }
    __syncthreads();
    if (t == 0) {
      float bb = red_d[0]; int ii = red_i[0];
      for (int w = 1; w < THREADS / 32; ++w)
        if (red_d[w] > bb || (red_d[w] == bb && red_i[w] < ii)) { bb = red_d[w]; ii = red_i[w]; }
      red_i[0] = ii;
    }
    __syncthreads();
    far = red_i[0];
    __syncthreads();
  }
}

// ---------------------------------------------------------------------------
// Ball query: one wave per query center; in-index-order ballot scan collects
// the 32 lowest-index points within radius (== reference sort-truncate).
// ---------------------------------------------------------------------------
__global__ void ball_query_kernel(const float* __restrict__ xyz, const float* __restrict__ new_xyz,
                                  int* __restrict__ gidx, int Npts, int M, float r2) {
  int gw = (blockIdx.x * blockDim.x + threadIdx.x) >> 5;
  int lane = threadIdx.x & 31;
  if (gw >= BATCH * M) return;
  int b = gw / M, m = gw % M;
  const float* px = xyz + (size_t)b * Npts * 3;
  const float* c  = new_xyz + ((size_t)b * M + m) * 3;
  float cx = c[0], cy = c[1], cz = c[2];
  int* out = gidx + ((size_t)b * M + m) * 32;
  int count = 0;
  int firstIdx = Npts - 1;
  bool haveFirst = false;
  for (int base = 0; base < Npts && count < 32; base += 32) {
    int i = base + lane;
    bool ok = false;
    if (i < Npts) {
      float dx = px[i * 3 + 0] - cx, dy = px[i * 3 + 1] - cy, dz = px[i * 3 + 2] - cz;
      ok = (dx * dx + dy * dy + dz * dz) <= r2;
    }
    unsigned mask = (unsigned)__ballot(ok);
    if (!haveFirst && mask) { firstIdx = base + __ffs(mask) - 1; haveFirst = true; }
    int rank = __popc(mask & ((1u << lane) - 1u));
    if (ok) { int slot = count + rank; if (slot < 32) out[slot] = i; }
    count += __popc(mask);
  }
  if (lane >= count) out[lane] = firstIdx;
}

// ---------------------------------------------------------------------------
// Build grouped features: [rel_xyz(3) | feats(C) | zero-pad to Kpad], f16.
// ---------------------------------------------------------------------------
__global__ void group_kernel(const float* __restrict__ xyz, const float* __restrict__ new_xyz,
                             const _Float16* __restrict__ feats, const int* __restrict__ gidx,
                             _Float16* __restrict__ grouped, int Npts, int M, int C, int Kpad) {
  int tid = blockIdx.x * blockDim.x + threadIdx.x;
  int total = BATCH * M * 32;
  if (tid >= total) return;
  int b = tid / (M * 32);
  int m = (tid >> 5) % M;
  int idx = gidx[tid];
  const float* p = xyz + ((size_t)b * Npts + idx) * 3;
  const float* c = new_xyz + ((size_t)b * M + m) * 3;
  _Float16* o = grouped + (size_t)tid * Kpad;
  o[0] = (_Float16)(p[0] - c[0]);
  o[1] = (_Float16)(p[1] - c[1]);
  o[2] = (_Float16)(p[2] - c[2]);
  int j = 3;
  if (feats) {
    const _Float16* f = feats + ((size_t)b * Npts + idx) * C;
    for (int cc = 0; cc < C; ++cc) o[j++] = f[cc];
  }
  for (; j < Kpad; ++j) o[j] = (_Float16)0.0f;
}

// ---------------------------------------------------------------------------
// WMMA GEMM: out[rows,C] = act(A[rows,K] @ W + bias). One wave per 16x16 tile,
// v_wmma_f32_16x16x32_f16 over K/32 steps.
//
// A (row-major activations), documented 16-bit A fragment layout:
//   lanes 0-15 hold row M=lane,   K = {k0..k0+7, k0+16..k0+23}
//   lanes16-31 hold row M=lane-16,K = {k0+8..15, k0+24..31}
//   -> two aligned 16B v8h loads per lane.
//
// B (weights) are PRE-PACKED into fragment order by the conversion kernel:
//   packed[((ntile*kBlocks + kblk)*32 + lane)*16 + h] =
//       W[kblk*32 + 8*(lane/16) + h%8 + 16*(h/8)][ntile*16 + lane%16]
//   -> one aligned 32B v16h load per lane per k-step, pointer induction only.
//
// C/D layout: acc[r] = C[tm + r + 8*(lane/16)][tn + lane%16].
// ---------------------------------------------------------------------------
template<bool RELU, bool F32OUT>
__global__ __launch_bounds__(256)
void gemm_wmma_kernel(const _Float16* __restrict__ A, const _Float16* __restrict__ Bp,
                      const float* __restrict__ bias, int Creal,
                      _Float16* __restrict__ outH, float* __restrict__ outF,
                      int rows, int K, int C) {
  const int waveId = (blockIdx.x * blockDim.x + threadIdx.x) >> 5;
  const int lane = threadIdx.x & 31;
  const int tilesN = C >> 4;
  const int tiles = (rows >> 4) * tilesN;
  if (waveId >= tiles) return;
  const int tm = (waveId / tilesN) << 4;
  const int tn = (waveId % tilesN) << 4;
  const int halfSel = lane >> 4;
  const int l16 = lane & 15;
  const int kBlocks = K >> 5;

  const _Float16* ap = A + (size_t)(tm + l16) * K + halfSel * 8;
  const _Float16* bp = Bp + (((size_t)(tn >> 4) * kBlocks) * 32 + lane) * 16;

  v8f acc = {};
  for (int kb = 0; kb < kBlocks; ++kb) {
    v8h alo = *(const v8h*)ap;
    v8h ahi = *(const v8h*)(ap + 16);
    v16h bfrag = *(const v16h*)bp;
    __builtin_prefetch(ap + 32, 0, 1);   // next A k-chunk -> global_prefetch_b8
    v16h a;
#pragma unroll
    for (int i = 0; i < 8; ++i) { a[i] = alo[i]; a[8 + i] = ahi[i]; }
    acc = __builtin_amdgcn_wmma_f32_16x16x32_f16(false, a, false, bfrag,
                                                 (short)0, acc, false, false);
    ap += 32;
    bp += 512; // 32 lanes * 16 halves
  }
  const int col = tn + l16;
  const float bv = (col < Creal) ? bias[col] : 0.0f;
#pragma unroll
  for (int r = 0; r < 8; ++r) {
    int row = tm + r + 8 * halfSel;
    float v = acc[r] + bv;
    if (RELU) v = fmaxf(v, 0.0f);
    if (F32OUT) outF[(size_t)row * C + col] = v;
    else        outH[(size_t)row * C + col] = (_Float16)v;
  }
}

// max over the 32 samples of each group
__global__ void maxpool_kernel(const _Float16* __restrict__ act, _Float16* __restrict__ out,
                               int groups, int C) {
  int tid = blockIdx.x * blockDim.x + threadIdx.x;
  if (tid >= groups * C) return;
  int g = tid / C, c = tid % C;
  const _Float16* a = act + ((size_t)g * 32) * C + c;
  float m = -1e30f;
  for (int s = 0; s < 32; ++s) m = fmaxf(m, (float)a[(size_t)s * C]);
  out[(size_t)g * C + c] = (_Float16)m;
}

// ---------------------------------------------------------------------------
// Feature propagation: 3-NN inverse-distance interpolation of pts2 onto xyz1,
// out row = [pts1(C1) | interp(C2)]. xyz2 staged in LDS (<=12KB).
// ---------------------------------------------------------------------------
__global__ __launch_bounds__(256)
void fp_interp_kernel(const float* __restrict__ xyz1, const float* __restrict__ xyz2,
                      const _Float16* __restrict__ pts2, const _Float16* __restrict__ pts1,
                      _Float16* __restrict__ out, int n1, int n2, int C2, int C1) {
  __shared__ float s2[1024 * 3];
  int b = blockIdx.y;
  int i = blockIdx.x * blockDim.x + threadIdx.x;
  for (int k = threadIdx.x; k < n2 * 3; k += blockDim.x)
    s2[k] = xyz2[(size_t)b * n2 * 3 + k];
  __syncthreads();
  if (i >= n1) return;
  const float* p = xyz1 + ((size_t)b * n1 + i) * 3;
  float x = p[0], y = p[1], z = p[2];
  float d0 = 1e30f, d1 = 1e30f, d2 = 1e30f;
  int i0 = 0, i1 = 0, i2 = 0;
  for (int j = 0; j < n2; ++j) {
    float dx = s2[j * 3 + 0] - x, dy = s2[j * 3 + 1] - y, dz = s2[j * 3 + 2] - z;
    float d = dx * dx + dy * dy + dz * dz;
    if (d < d0)      { d2 = d1; i2 = i1; d1 = d0; i1 = i0; d0 = d; i0 = j; }
    else if (d < d1) { d2 = d1; i2 = i1; d1 = d;  i1 = j; }
    else if (d < d2) { d2 = d;  i2 = j; }
  }
  float w0 = 1.0f / (d0 + 1e-8f), w1 = 1.0f / (d1 + 1e-8f), w2 = 1.0f / (d2 + 1e-8f);
  float wsum = w0 + w1 + w2;
  w0 /= wsum; w1 /= wsum; w2 /= wsum;
  int Cout = C1 + C2;
  _Float16* o = out + ((size_t)b * n1 + i) * Cout;
  if (pts1) {
    const _Float16* p1 = pts1 + ((size_t)b * n1 + i) * C1;
    for (int c = 0; c < C1; ++c) o[c] = p1[c];
  }
  const _Float16* f0 = pts2 + ((size_t)b * n2 + i0) * C2;
  const _Float16* f1 = pts2 + ((size_t)b * n2 + i1) * C2;
  const _Float16* f2 = pts2 + ((size_t)b * n2 + i2) * C2;
  for (int c = 0; c < C2; ++c)
    o[C1 + c] = (_Float16)(w0 * (float)f0[c] + w1 * (float)f1[c] + w2 * (float)f2[c]);
}

// ---------------------------------------------------------------------------
// fp32 W (cin x cout) -> f16, padded to (Kpad x Cpad) AND pre-swizzled into
// WMMA B-fragment order (see gemm_wmma_kernel header comment).
// ---------------------------------------------------------------------------
__global__ void convert_weight_pack_kernel(const float* __restrict__ W, _Float16* __restrict__ out,
                                           int cin, int cout, int Kpad, int Cpad) {
  int tid = blockIdx.x * blockDim.x + threadIdx.x;
  if (tid >= Kpad * Cpad) return;
  int h    = tid & 15;
  int lane = (tid >> 4) & 31;
  int rem  = tid >> 9;               // / (16*32)
  int kBlocks = Kpad >> 5;
  int kb = rem % kBlocks;
  int nb = rem / kBlocks;
  int k = (kb << 5) + ((lane >> 4) << 3) + (h & 7) + ((h >> 3) << 4);
  int c = (nb << 4) + (lane & 15);
  out[tid] = (_Float16)((k < cin && c < cout) ? W[k * cout + c] : 0.0f);
}

__global__ void copy_logits_kernel(const float* __restrict__ in, float* __restrict__ out, int rows) {
  int tid = blockIdx.x * blockDim.x + threadIdx.x;
  if (tid >= rows * 13) return;
  int r = tid / 13, c = tid % 13;
  out[tid] = in[(size_t)r * 16 + c];
}

// ---------------------------------------------------------------------------
struct WSpec { int widx, cin, cout, kpad, cpad; };
static const WSpec g_wspec[18] = {
  {1, 3, 32, 32, 32},   {3, 32, 32, 32, 32},  {5, 32, 64, 32, 64},     // sa1
  {7, 67, 64, 96, 64},  {9, 64, 64, 64, 64},  {11, 64, 128, 64, 128},  // sa2
  {13, 131, 128, 160, 128}, {15, 128, 128, 128, 128}, {17, 128, 256, 128, 256}, // sa3
  {19, 384, 256, 384, 256}, {21, 256, 128, 256, 128},                  // fp3
  {23, 192, 128, 192, 128}, {25, 128, 64, 128, 64},                    // fp2
  {27, 64, 64, 64, 64}, {29, 64, 64, 64, 64}, {31, 64, 64, 64, 64},    // fp1
  {33, 64, 64, 64, 64}, {35, 64, 13, 64, 16},                          // cls
};

extern "C" void kernel_launch(void* const* d_in, const int* in_sizes, int n_in,
                              void* d_out, int out_size, void* d_ws, size_t ws_size,
                              hipStream_t stream) {
  (void)in_sizes; (void)n_in; (void)out_size; (void)ws_size;
  const float* xyz = (const float*)d_in[0];
  float* out = (float*)d_out;

  // ---- workspace bump allocator -------------------------------------------
  char* base = (char*)d_ws;
  size_t off = 0;
  auto alloc = [&](size_t bytes) -> void* {
    void* p = base + off;
    off += (bytes + 255) & ~(size_t)255;
    return p;
  };
  float* l1_xyz = (float*)alloc((size_t)BATCH * 1024 * 3 * 4);
  float* l2_xyz = (float*)alloc((size_t)BATCH * 256 * 3 * 4);
  float* l3_xyz = (float*)alloc((size_t)BATCH * 64 * 3 * 4);
  int* gidx1 = (int*)alloc((size_t)BATCH * 1024 * 32 * 4);
  int* gidx2 = (int*)alloc((size_t)BATCH * 256 * 32 * 4);
  int* gidx3 = (int*)alloc((size_t)BATCH * 64 * 32 * 4);
  float* logits = (float*)alloc((size_t)BATCH * NPTS * 16 * 4);

  size_t woff[18]; size_t wtot = 0;
  for (int i = 0; i < 18; ++i) { woff[i] = wtot; wtot += (size_t)g_wspec[i].kpad * g_wspec[i].cpad; }
  _Float16* wts    = (_Float16*)alloc(wtot * 2);
  _Float16* arenaA = (_Float16*)alloc((size_t)BATCH * 1024 * 32 * 32 * 2); // 4.19M halves
  _Float16* P1     = (_Float16*)alloc((size_t)131072 * 64 * 2);            // 8.39M halves
  _Float16* P2     = (_Float16*)alloc((size_t)131072 * 64 * 2);
  _Float16* l1_p   = (_Float16*)alloc((size_t)BATCH * 1024 * 64 * 2);
  _Float16* l2_p   = (_Float16*)alloc((size_t)BATCH * 256 * 128 * 2);
  _Float16* l3_p   = (_Float16*)alloc((size_t)BATCH * 64 * 256 * 2);

  auto bias = [&](int i) { return (const float*)d_in[i]; };
  auto gemm = [&](const _Float16* A, int wi, const float* bv, int Creal,
                  _Float16* oH, float* oF, int rows, int K, int C, bool relu, bool f32out) {
    int tiles = (rows >> 4) * (C >> 4);
    int blocks = (tiles + 7) / 8; // 8 waves / 256-thread block
    if (relu)
      gemm_wmma_kernel<true, false><<<blocks, 256, 0, stream>>>(A, wts + woff[wi], bv, Creal, oH, nullptr, rows, K, C);
    else if (!f32out)
      gemm_wmma_kernel<false, false><<<blocks, 256, 0, stream>>>(A, wts + woff[wi], bv, Creal, oH, nullptr, rows, K, C);
    else
      gemm_wmma_kernel<false, true><<<blocks, 256, 0, stream>>>(A, wts + woff[wi], bv, Creal, nullptr, oF, rows, K, C);
  };

  // ---- weight conversion + fragment packing --------------------------------
  for (int i = 0; i < 18; ++i) {
    int total = g_wspec[i].kpad * g_wspec[i].cpad;
    convert_weight_pack_kernel<<<(total + 255) / 256, 256, 0, stream>>>(
        (const float*)d_in[g_wspec[i].widx], wts + woff[i],
        g_wspec[i].cin, g_wspec[i].cout, g_wspec[i].kpad, g_wspec[i].cpad);
  }

  // ---- SA1: 16384 -> 1024, r=0.1, MLP 3->32->32->64 -----------------------
  fps_kernel<16384, 1024, 1024><<<BATCH, 1024, 0, stream>>>(xyz, l1_xyz);
  ball_query_kernel<<<(BATCH * 1024 * 32 + 255) / 256, 256, 0, stream>>>(xyz, l1_xyz, gidx1, 16384, 1024, 0.01f);
  group_kernel<<<(BATCH * 1024 * 32 + 255) / 256, 256, 0, stream>>>(xyz, l1_xyz, (const _Float16*)nullptr, gidx1, arenaA, 16384, 1024, 0, 32);
  gemm(arenaA, 0, bias(2), 32, P1, nullptr, 131072, 32, 32, true, false);
  gemm(P1,     1, bias(4), 32, P2, nullptr, 131072, 32, 32, true, false);
  gemm(P2,     2, bias(6), 64, P1, nullptr, 131072, 32, 64, true, false);
  maxpool_kernel<<<(4096 * 64 + 255) / 256, 256, 0, stream>>>(P1, l1_p, 4096, 64);

  // ---- SA2: 1024 -> 256, r=0.2, MLP 67->64->64->128 -----------------------
  fps_kernel<1024, 256, 256><<<BATCH, 256, 0, stream>>>(l1_xyz, l2_xyz);
  ball_query_kernel<<<(BATCH * 256 * 32 + 255) / 256, 256, 0, stream>>>(l1_xyz, l2_xyz, gidx2, 1024, 256, 0.04f);
  group_kernel<<<(BATCH * 256 * 32 + 255) / 256, 256, 0, stream>>>(l1_xyz, l2_xyz, l1_p, gidx2, arenaA, 1024, 256, 64, 96);
  gemm(arenaA, 3, bias(8),  64,  P1, nullptr, 32768, 96, 64, true, false);
  gemm(P1,     4, bias(10), 64,  P2, nullptr, 32768, 64, 64, true, false);
  gemm(P2,     5, bias(12), 128, P1, nullptr, 32768, 64, 128, true, false);
  maxpool_kernel<<<(1024 * 128 + 255) / 256, 256, 0, stream>>>(P1, l2_p, 1024, 128);

  // ---- SA3: 256 -> 64, r=0.4, MLP 131->128->128->256 ----------------------
  fps_kernel<256, 64, 256><<<BATCH, 256, 0, stream>>>(l2_xyz, l3_xyz);
  ball_query_kernel<<<(BATCH * 64 * 32 + 255) / 256, 256, 0, stream>>>(l2_xyz, l3_xyz, gidx3, 256, 64, 0.16f);
  group_kernel<<<(BATCH * 64 * 32 + 255) / 256, 256, 0, stream>>>(l2_xyz, l3_xyz, l2_p, gidx3, arenaA, 256, 64, 128, 160);
  gemm(arenaA, 6, bias(14), 128, P1, nullptr, 8192, 160, 128, true, false);
  gemm(P1,     7, bias(16), 128, P2, nullptr, 8192, 128, 128, true, false);
  gemm(P2,     8, bias(18), 256, P1, nullptr, 8192, 128, 256, true, false);
  maxpool_kernel<<<(256 * 256 + 255) / 256, 256, 0, stream>>>(P1, l3_p, 256, 256);

  // ---- FP3: interp l3 -> l2, MLP 384->256->128 ----------------------------
  fp_interp_kernel<<<dim3(1, BATCH), 256, 0, stream>>>(l2_xyz, l3_xyz, l3_p, l2_p, arenaA, 256, 64, 256, 128);
  gemm(arenaA, 9,  bias(20), 256, P1,   nullptr, 1024, 384, 256, true, false);
  gemm(P1,     10, bias(22), 128, l2_p, nullptr, 1024, 256, 128, true, false);

  // ---- FP2: interp l2 -> l1, MLP 192->128->64 -----------------------------
  fp_interp_kernel<<<dim3(4, BATCH), 256, 0, stream>>>(l1_xyz, l2_xyz, l2_p, l1_p, arenaA, 1024, 256, 128, 64);
  gemm(arenaA, 11, bias(24), 128, P1,   nullptr, 4096, 192, 128, true, false);
  gemm(P1,     12, bias(26), 64,  l1_p, nullptr, 4096, 128, 64, true, false);

  // ---- FP1: interp l1 -> l0, MLP 64->64->64->64 ---------------------------
  fp_interp_kernel<<<dim3(64, BATCH), 256, 0, stream>>>(xyz, l1_xyz, l1_p, (const _Float16*)nullptr, arenaA, 16384, 1024, 64, 0);
  gemm(arenaA, 13, bias(28), 64, P1, nullptr, 65536, 64, 64, true, false);
  gemm(P1,     14, bias(30), 64, P2, nullptr, 65536, 64, 64, true, false);
  gemm(P2,     15, bias(32), 64, P1, nullptr, 65536, 64, 64, true, false);

  // ---- classifier (no relu): 64->64->13 (padded to 16) --------------------
  gemm(P1, 16, bias(34), 64, P2, nullptr, 65536, 64, 64, false, false);
  gemm(P2, 17, bias(36), 13, nullptr, logits, 65536, 64, 16, false, true);
  copy_logits_kernel<<<(65536 * 13 + 255) / 256, 256, 0, stream>>>(logits, out, 65536);
}
